// MultiHeadedAttention_5162550690137
// MI455X (gfx1250) — matrix operand order repaired and checked by reading, not compiled, not run on hardware
//
#include <hip/hip_runtime.h>

// MI455X (gfx1250) relative-position MHA.
// Compute-bound (~103 GFLOP vs ~7us HBM floor at 23.3TB/s): all matmuls via
// v_wmma_f32_16x16x32_bf16 (fp32 accumulate), wave32. Global->LDS staging is
// pipelined with gfx1250 async-to-LDS ops (ASYNCcnt) where available.

typedef __bf16 bf16;
typedef __attribute__((ext_vector_type(16))) __bf16 v16bf;
typedef __attribute__((ext_vector_type(8)))  __bf16 v8bf;
typedef __attribute__((ext_vector_type(4)))  __bf16 v4bf;
typedef __attribute__((ext_vector_type(8)))  float  v8f;
typedef __attribute__((ext_vector_type(4)))  int    v4i;

static constexpr int Bc  = 2;
static constexpr int Lc  = 2048;
static constexpr int Dc  = 1024;
static constexpr int Hc  = 16;
static constexpr int KCLIP = 32;
static constexpr int DKc = 64;
static constexpr int Mrows = Bc * Lc;       // 4096
static constexpr int NREL  = 2 * KCLIP + 1; // 65
static constexpr int RPAD  = 80;            // rel rows padded (5 x 16), Qrel row width

// ---------------- gfx1250 async global->LDS (guarded; sync fallback) ----------
#if __has_builtin(__builtin_amdgcn_global_load_async_to_lds_b128)
#define HAS_ASYNC_LDS 1
#else
#define HAS_ASYNC_LDS 0
#endif

__device__ __forceinline__ void async_copy16(const void* g, void* l) {
#if HAS_ASYNC_LDS
  __builtin_amdgcn_global_load_async_to_lds_b128(
      (__attribute__((address_space(1))) v4i*)(g),
      (__attribute__((address_space(3))) v4i*)(l), 0, 0);
#else
  *(uint4*)l = *(const uint4*)g;
#endif
}

#if HAS_ASYNC_LDS
#if __has_builtin(__builtin_amdgcn_s_wait_asynccnt)
#define WAIT_ASYNC(n) __builtin_amdgcn_s_wait_asynccnt(n)
#else
#define WAIT_ASYNC(n) asm volatile("s_wait_asynccnt %0" ::"i"(n) : "memory")
#endif
#else
#define WAIT_ASYNC(n) ((void)0)
#endif

// ------------------------------------------------------------------ helpers
__device__ __forceinline__ v8f vzero8() {
  v8f v;
#pragma unroll
  for (int i = 0; i < 8; ++i) v[i] = 0.f;
  return v;
}

__device__ __forceinline__ v8f wmma_bf16(v16bf a, v16bf b, v8f c) {
  return __builtin_amdgcn_wmma_f32_16x16x32_bf16(false, a, false, b, (short)0, c,
                                                 false, false);
}

// A-fragment (16x32 bf16, row-major source, row stride ld elements):
// lane: m = lane&15, half = lane>>4; elems 0..7 -> k = half*8+e ; 8..15 -> k = 16+half*8+e
__device__ __forceinline__ v16bf load_fragA(const bf16* p, int row0, int k0, int ld) {
  const int lane = threadIdx.x & 31;
  const bf16* r = p + (size_t)(row0 + (lane & 15)) * ld + k0 + ((lane >> 4) * 8);
  union { v16bf v; v8bf h[2]; } u;
  u.h[0] = *(const v8bf*)r;
  u.h[1] = *(const v8bf*)(r + 16);
  return u.v;
}

// B-fragment (32x16 bf16) from an n-major staging buffer Bt[n][k] (stride ld):
// lane: n = lane&15, half = lane>>4; elems 0..7 -> k = half*8+e ; 8..15 -> k = 16+half*8+e
__device__ __forceinline__ v16bf load_fragB(const bf16* p, int n0, int k0, int ld) {
  const int lane = threadIdx.x & 31;
  const bf16* r = p + (size_t)(n0 + (lane & 15)) * ld + k0 + ((lane >> 4) * 8);
  union { v16bf v; v8bf h[2]; } u;
  u.h[0] = *(const v8bf*)r;
  u.h[1] = *(const v8bf*)(r + 16);
  return u.v;
}

// ---------------------------------------------------------------- converts
__global__ void cvt_f32_to_bf16(const float* __restrict__ s, bf16* __restrict__ d, int n4) {
  int i = blockIdx.x * blockDim.x + threadIdx.x;
  if (i >= n4) return;
  float4 f = ((const float4*)s)[i];
  v4bf o;
  o[0] = (bf16)f.x; o[1] = (bf16)f.y; o[2] = (bf16)f.z; o[3] = (bf16)f.w;
  ((v4bf*)d)[i] = o;
}

__global__ void pad_rel_kernel(const float* __restrict__ rel, bf16* __restrict__ out) {
  int idx = blockIdx.x * blockDim.x + threadIdx.x;
  if (idx >= RPAD * DKc) return;
  int r = idx >> 6, c = idx & 63;
  out[idx] = (r < NREL) ? (bf16)rel[r * DKc + c] : (bf16)0.f;
}

// ---------------------------------------------------------------- GEMM
// C = A[M,K] * B[K,N] + bias ; MODE 0: bf16 out scattered to [B,H,L,DK] (with scale)
//                              MODE 1: f32 out row-major [M,N]
// A tile is double-buffered and staged with async-to-LDS one iteration ahead.
template <int MODE>
__global__ __launch_bounds__(256)
void gemm_bf16_kernel(const bf16* __restrict__ A, const bf16* __restrict__ Bw,
                      const float* __restrict__ bias, float* __restrict__ outF,
                      bf16* __restrict__ outH, int M, int N, int Kd, float scale) {
  __shared__ bf16 sA[2][128][64];  // m-major, double buffered
  __shared__ bf16 sB[128][64];     // n-major (transposed W tile)

  const int tid = threadIdx.x;
  const int w = tid >> 5, wm = w >> 1, wn = w & 1;
  const int m0 = blockIdx.x * 128;
  const int n0 = blockIdx.y * 128;

  v8f acc[2][4];
#pragma unroll
  for (int a = 0; a < 2; ++a)
#pragma unroll
    for (int b = 0; b < 4; ++b) acc[a][b] = vzero8();

  const int arow = tid >> 3, acol = (tid & 7) * 8;  // 32 rows / iter step
  // prologue: stage A tile 0
#pragma unroll
  for (int it = 0; it < 4; ++it)
    async_copy16(&A[(size_t)(m0 + arow + it * 32) * Kd + acol],
                 &sA[0][arow + it * 32][acol]);

  const int nIter = Kd >> 6;
  for (int t = 0; t < nIter; ++t) {
    const int k0 = t << 6;
    if (t + 1 < nIter) {  // stage A tile t+1 (async, overlaps compute of t)
#pragma unroll
      for (int it = 0; it < 4; ++it)
        async_copy16(&A[(size_t)(m0 + arow + it * 32) * Kd + (k0 + 64) + acol],
                     &sA[(t + 1) & 1][arow + it * 32][acol]);
    }
#pragma unroll
    for (int it = 0; it < 4; ++it) {  // B tile: 64x128 -> transposed store
      int idx = tid + it * 256;
      int kr = idx >> 4, c = (idx & 15) * 8;
      union { uint4 u; bf16 e[8]; } tt;
      tt.u = *(const uint4*)&Bw[(size_t)(k0 + kr) * N + n0 + c];
#pragma unroll
      for (int e = 0; e < 8; ++e) sB[c + e][kr] = tt.e[e];
    }
    if (t + 1 < nIter) { WAIT_ASYNC(4); } else { WAIT_ASYNC(0); }
    __syncthreads();

    const bf16* curA = &sA[t & 1][0][0];
#pragma unroll
    for (int kk = 0; kk < 64; kk += 32) {
      v16bf am0 = load_fragA(curA, wm * 32, kk, 64);
      v16bf am1 = load_fragA(curA, wm * 32 + 16, kk, 64);
#pragma unroll
      for (int nt = 0; nt < 4; ++nt) {
        v16bf bn = load_fragB(&sB[0][0], wn * 64 + nt * 16, kk, 64);
        acc[0][nt] = wmma_bf16(am0, bn, acc[0][nt]);
        acc[1][nt] = wmma_bf16(am1, bn, acc[1][nt]);
      }
    }
    __syncthreads();
  }

  const int lane = tid & 31, nlo = lane & 15, half = lane >> 4;
#pragma unroll
  for (int mt = 0; mt < 2; ++mt)
#pragma unroll
    for (int nt = 0; nt < 4; ++nt) {
      int gn = n0 + wn * 64 + nt * 16 + nlo;
      float bv = bias[gn];
#pragma unroll
      for (int r = 0; r < 8; ++r) {
        int gm = m0 + wm * 32 + mt * 16 + r + half * 8;
        float v = (acc[mt][nt][r] + bv) * scale;
        if (MODE == 0) {
          int bb = gm >> 11, ii = gm & (Lc - 1);   // L = 2048
          int hh = gn >> 6, dd = gn & (DKc - 1);   // DK = 64
          outH[(((size_t)(bb * Hc + hh) * Lc + ii) << 6) + dd] = (bf16)v;
        } else {
          outF[(size_t)gm * N + gn] = v;
        }
      }
    }
}

// ---------------------------------------------------------------- Qrel = Q @ rel^T
// Q: [B,H,L,64] bf16 (already scaled) ; relbf: [80][64] ; Qrel: [B*H*L, 80] f32
__global__ __launch_bounds__(256)
void qrel_kernel(const bf16* __restrict__ Q, const bf16* __restrict__ relbf,
                 float* __restrict__ Qrel) {
  const int w = threadIdx.x >> 5;
  const int row0 = blockIdx.x * 128 + w * 16;
  const int lane = threadIdx.x & 31, nlo = lane & 15, half = lane >> 4;
  // batch all fragment loads so WMMAs issue under a single wait
  v16bf a0 = load_fragA(Q, row0, 0, DKc);
  v16bf a1 = load_fragA(Q, row0, 32, DKc);
  v16bf b0[5], b1[5];
#pragma unroll
  for (int nt = 0; nt < 5; ++nt) {
    b0[nt] = load_fragB(relbf, nt * 16, 0, DKc);
    b1[nt] = load_fragB(relbf, nt * 16, 32, DKc);
  }
  v8f c[5];
#pragma unroll
  for (int nt = 0; nt < 5; ++nt) c[nt] = wmma_bf16(a0, b0[nt], vzero8());
#pragma unroll
  for (int nt = 0; nt < 5; ++nt) c[nt] = wmma_bf16(a1, b1[nt], c[nt]);
#pragma unroll
  for (int nt = 0; nt < 5; ++nt)
#pragma unroll
    for (int r = 0; r < 8; ++r)
      Qrel[(size_t)(row0 + r + half * 8) * RPAD + nt * 16 + nlo] = c[nt][r];
}

// ---------------------------------------------------------------- flash attention
// grid.x = B*H*(L/128); block 256 = 8 waves, each wave owns a 16-row i tile.
__global__ __launch_bounds__(256)
void attn_kernel(const bf16* __restrict__ Q, const bf16* __restrict__ Kh,
                 const bf16* __restrict__ Vh, const float* __restrict__ Qrel,
                 const bf16* __restrict__ relbf, const unsigned char* __restrict__ mask,
                 bf16* __restrict__ ctx) {
  __shared__ float sArel[8][16][66];  // per-wave rel-bucket mass (f32, LDS atomics)
  __shared__ float sScale[8][16];
  __shared__ float sRel64[64];        // rel_emb row r=64 (rank-1 epilogue term)
  __shared__ bf16  sRelT[64][64];     // rel^T for buckets 0..63
  __shared__ bf16  sK[2][32][64];     // K tile, j-major, double buffered (async)
  __shared__ bf16  sVt[64][32];       // V tile transposed, dk-major
  __shared__ bf16  sP[8][16][32];     // per-wave P staging (A-fragment source)

  const int tid = threadIdx.x, w = tid >> 5, lane = tid & 31;
  const int nlo = lane & 15, half = lane >> 4;
  const int gid = blockIdx.x;
  const int ib = gid & 15, h = (gid >> 4) & (Hc - 1), b = gid >> 8;

  const size_t headoff = (size_t)(b * Hc + h) * Lc * DKc;
  const bf16* Qh = Q + headoff;
  const bf16* Kp = Kh + headoff;
  const bf16* Vp = Vh + headoff;
  const float* qrl = Qrel + (size_t)(b * Hc + h) * Lc * RPAD;
  const unsigned char* mk = mask + (size_t)b * Lc * Lc;

  for (int idx = tid; idx < 64 * 64; idx += 256) {
    int dk = idx >> 6, r = idx & 63;
    sRelT[dk][r] = relbf[r * DKc + dk];
  }
  if (tid < 64) sRel64[tid] = (float)relbf[64 * DKc + tid];
  for (int idx = tid; idx < 8 * 16 * 66; idx += 256) ((float*)sArel)[idx] = 0.f;

  const int krow = tid >> 3, kcol = (tid & 7) * 8;
  async_copy16(&Kp[(size_t)krow * DKc + kcol], &sK[0][krow][kcol]);  // K tile 0
  __syncthreads();

  const int i0 = ib * 128 + w * 16;
  const v16bf aq0 = load_fragA(Qh, i0, 0, DKc);
  const v16bf aq1 = load_fragA(Qh, i0, 32, DKc);

  v8f acc[4];
#pragma unroll
  for (int nt = 0; nt < 4; ++nt) acc[nt] = vzero8();

  float mrow[8], lrow[8];
  int irow[8];
#pragma unroll
  for (int r = 0; r < 8; ++r) {
    mrow[r] = -3.0e38f;
    lrow[r] = 0.f;
    irow[r] = i0 + r + half * 8;
  }

  const int nTiles = Lc / 32;
  for (int t = 0; t < nTiles; ++t) {
    const int jt = t * 32;
    if (t + 1 < nTiles) {  // async K(t+1), prefetch V(t+1)
      async_copy16(&Kp[(size_t)(jt + 32 + krow) * DKc + kcol],
                   &sK[(t + 1) & 1][krow][kcol]);
      __builtin_prefetch(&Vp[(size_t)(jt + 32 + krow) * DKc + kcol], 0, 1);
    }
    {  // V^T staging (one uint4 = 8 bf16 per thread, scattered bf16 stores)
      union { uint4 u; bf16 e[8]; } tt;
      tt.u = *(const uint4*)&Vp[(size_t)(jt + krow) * DKc + kcol];
#pragma unroll
      for (int e = 0; e < 8; ++e) sVt[kcol + e][krow] = tt.e[e];
    }
    if (t + 1 < nTiles) { WAIT_ASYNC(1); } else { WAIT_ASYNC(0); }
    __syncthreads();

    const bf16* curK = &sK[t & 1][0][0];
    // S = q K^T for two 16-wide j tiles
    v8f s0 = vzero8(), s1 = vzero8();
    s0 = wmma_bf16(aq0, load_fragB(curK, 0, 0, 64), s0);
    s0 = wmma_bf16(aq1, load_fragB(curK, 0, 32, 64), s0);
    s1 = wmma_bf16(aq0, load_fragB(curK, 16, 0, 64), s1);
    s1 = wmma_bf16(aq1, load_fragB(curK, 16, 32, 64), s1);

    const int j0 = jt + nlo, j1 = jt + 16 + nlo;
    int rel0[8], rel1[8];
#pragma unroll
    for (int r = 0; r < 8; ++r) {  // + relative-key term, + mask
      int i = irow[r];
      int d0 = j0 - i; d0 = (d0 < -KCLIP) ? -KCLIP : (d0 > KCLIP ? KCLIP : d0); d0 += KCLIP;
      int d1 = j1 - i; d1 = (d1 < -KCLIP) ? -KCLIP : (d1 > KCLIP ? KCLIP : d1); d1 += KCLIP;
      rel0[r] = d0; rel1[r] = d1;
      float v0 = s0[r] + qrl[(size_t)i * RPAD + d0];
      float v1 = s1[r] + qrl[(size_t)i * RPAD + d1];
      if (!mk[(size_t)i * Lc + j0]) v0 = -1e18f;
      if (!mk[(size_t)i * Lc + j1]) v1 = -1e18f;
      s0[r] = v0; s1[r] = v1;
    }

    // online softmax (row stats reduced across the 16 lanes of each half)
    float p0a[8], p1a[8], scl[8];
    bool resc = false;
#pragma unroll
    for (int r = 0; r < 8; ++r) {
      float tm = fmaxf(s0[r], s1[r]);
      tm = fmaxf(tm, __shfl_xor(tm, 1, 32));
      tm = fmaxf(tm, __shfl_xor(tm, 2, 32));
      tm = fmaxf(tm, __shfl_xor(tm, 4, 32));
      tm = fmaxf(tm, __shfl_xor(tm, 8, 32));
      float mnew = fmaxf(mrow[r], tm);
      float sc = __expf(mrow[r] - mnew);
      mrow[r] = mnew;
      float p0 = __expf(s0[r] - mnew);
      float p1 = __expf(s1[r] - mnew);
      float ps = p0 + p1;
      ps += __shfl_xor(ps, 1, 32);
      ps += __shfl_xor(ps, 2, 32);
      ps += __shfl_xor(ps, 4, 32);
      ps += __shfl_xor(ps, 8, 32);
      lrow[r] = lrow[r] * sc + ps;
      scl[r] = sc;
      resc |= (sc != 1.0f);
      p0a[r] = p0; p1a[r] = p1;
#pragma unroll
      for (int nt = 0; nt < 4; ++nt) acc[nt][r] *= sc;
    }

    // stage P (bf16) for the PV WMMA
#pragma unroll
    for (int r = 0; r < 8; ++r) {
      sP[w][r + half * 8][nlo] = (bf16)p0a[r];
      sP[w][r + half * 8][16 + nlo] = (bf16)p1a[r];
    }

    // rel-value bucket mass: rescale (wave-voted) then LDS atomic adds
    if (__any((int)resc)) {
#pragma unroll
      for (int r = 0; r < 8; ++r)
        if (nlo == r) sScale[w][r + half * 8] = scl[r];
      __builtin_amdgcn_wave_barrier();
      for (int idx = lane; idx < 16 * 66; idx += 32)
        ((float*)sArel[w])[idx] *= sScale[w][idx / 66];
    }
#pragma unroll
    for (int r = 0; r < 8; ++r) {
      atomicAdd(&sArel[w][r + half * 8][rel0[r]], p0a[r]);
      atomicAdd(&sArel[w][r + half * 8][rel1[r]], p1a[r]);
    }

    // O += P V
    v16bf ap = load_fragA(&sP[w][0][0], 0, 0, 32);
#pragma unroll
    for (int nt = 0; nt < 4; ++nt) {
      v16bf bv = load_fragB(&sVt[0][0], nt * 16, 0, 32);
      acc[nt] = wmma_bf16(ap, bv, acc[nt]);
    }
    __syncthreads();
  }

  // epilogue: O += Arel @ rel_emb. Buckets 0..63 via WMMA, bucket 64 rank-1.
#pragma unroll
  for (int kc = 0; kc < 2; ++kc) {
    v16bf aa;
#pragma unroll
    for (int e = 0; e < 8; ++e) {
      int kA = kc * 32 + half * 8 + e;
      int kB = kc * 32 + 16 + half * 8 + e;
      aa[e]     = (bf16)sArel[w][nlo][kA];
      aa[8 + e] = (bf16)sArel[w][nlo][kB];
    }
#pragma unroll
    for (int nt = 0; nt < 4; ++nt) {
      v16bf br = load_fragB(&sRelT[0][0], nt * 16, kc * 32, 64);
      acc[nt] = wmma_bf16(aa, br, acc[nt]);
    }
  }
  float a64[8];
#pragma unroll
  for (int r = 0; r < 8; ++r) a64[r] = sArel[w][r + half * 8][64];
#pragma unroll
  for (int nt = 0; nt < 4; ++nt) {
    float rv = sRel64[nt * 16 + nlo];
#pragma unroll
    for (int r = 0; r < 8; ++r) acc[nt][r] += a64[r] * rv;
  }

  // normalize and store ctx as [B, L, H*DK] bf16 (feeds the output projection GEMM)
#pragma unroll
  for (int r = 0; r < 8; ++r) {
    float inv = (lrow[r] > 0.f) ? 1.0f / lrow[r] : 0.f;
    size_t base = ((size_t)(b * Lc + irow[r])) * Dc + h * DKc;
#pragma unroll
    for (int nt = 0; nt < 4; ++nt)
      ctx[base + nt * 16 + nlo] = (bf16)(acc[nt][r] * inv);
  }
}

// ---------------------------------------------------------------- host
extern "C" void kernel_launch(void* const* d_in, const int* in_sizes, int n_in,
                              void* d_out, int out_size, void* d_ws, size_t ws_size,
                              hipStream_t stream) {
  const float* q_in = (const float*)d_in[0];
  const float* k_in = (const float*)d_in[1];
  const float* v_in = (const float*)d_in[2];
  const unsigned char* mask = (const unsigned char*)d_in[3];
  const float* Wq = (const float*)d_in[4];
  const float* bq = (const float*)d_in[5];
  const float* Wk = (const float*)d_in[6];
  const float* bk = (const float*)d_in[7];
  const float* Wv = (const float*)d_in[8];
  const float* bv = (const float*)d_in[9];
  const float* Wo = (const float*)d_in[10];
  const float* bo = (const float*)d_in[11];
  const float* rel = (const float*)d_in[12];
  float* out = (float*)d_out;

  const size_t SZ_XBF  = (size_t)Mrows * Dc * sizeof(bf16);          // 8 MB
  const size_t SZ_WBF  = (size_t)Dc * Dc * sizeof(bf16);             // 2 MB
  const size_t SZ_REL  = (size_t)RPAD * DKc * sizeof(bf16);
  const size_t SZ_HEAD = (size_t)Bc * Hc * Lc * DKc * sizeof(bf16);  // 8 MB
  const size_t SZ_QREL = (size_t)Bc * Hc * Lc * RPAD * sizeof(float);
  const size_t SZ_CTX  = (size_t)Mrows * Dc * sizeof(bf16);

  char* ws = (char*)d_ws;
  size_t off = 0;
  auto carve = [&](size_t bytes) {
    char* p = ws + off;
    off = (off + bytes + 255) & ~(size_t)255;
    return p;
  };
  bf16* xq = (bf16*)carve(SZ_XBF);
  bf16* xk = (bf16*)carve(SZ_XBF);
  bf16* xv = (bf16*)carve(SZ_XBF);
  bf16* wq = (bf16*)carve(SZ_WBF);
  bf16* wk = (bf16*)carve(SZ_WBF);
  bf16* wv = (bf16*)carve(SZ_WBF);
  bf16* wo = (bf16*)carve(SZ_WBF);
  bf16* relbf = (bf16*)carve(SZ_REL);
  bf16* qh = (bf16*)carve(SZ_HEAD);
  bf16* kh = (bf16*)carve(SZ_HEAD);
  bf16* vh = (bf16*)carve(SZ_HEAD);
  float* qrel = (float*)carve(SZ_QREL);
  bf16* ctx = (bf16*)carve(SZ_CTX);
  (void)ws_size; (void)n_in; (void)in_sizes; (void)out_size;

  const int nX4 = Mrows * Dc / 4;  // 1,048,576
  const int nW4 = Dc * Dc / 4;     //   262,144
  cvt_f32_to_bf16<<<(nX4 + 255) / 256, 256, 0, stream>>>(q_in, xq, nX4);
  cvt_f32_to_bf16<<<(nX4 + 255) / 256, 256, 0, stream>>>(k_in, xk, nX4);
  cvt_f32_to_bf16<<<(nX4 + 255) / 256, 256, 0, stream>>>(v_in, xv, nX4);
  cvt_f32_to_bf16<<<(nW4 + 255) / 256, 256, 0, stream>>>(Wq, wq, nW4);
  cvt_f32_to_bf16<<<(nW4 + 255) / 256, 256, 0, stream>>>(Wk, wk, nW4);
  cvt_f32_to_bf16<<<(nW4 + 255) / 256, 256, 0, stream>>>(Wv, wv, nW4);
  cvt_f32_to_bf16<<<(nW4 + 255) / 256, 256, 0, stream>>>(Wo, wo, nW4);
  pad_rel_kernel<<<(RPAD * DKc + 255) / 256, 256, 0, stream>>>(rel, relbf);

  dim3 ggrid(Mrows / 128, Dc / 128);  // (32, 8)
  const float qscale = 0.125f;        // 1/sqrt(64), folded into Q projection
  gemm_bf16_kernel<0><<<ggrid, 256, 0, stream>>>(xq, wq, bq, nullptr, qh,
                                                 Mrows, Dc, Dc, qscale);
  gemm_bf16_kernel<0><<<ggrid, 256, 0, stream>>>(xk, wk, bk, nullptr, kh,
                                                 Mrows, Dc, Dc, 1.0f);
  gemm_bf16_kernel<0><<<ggrid, 256, 0, stream>>>(xv, wv, bv, nullptr, vh,
                                                 Mrows, Dc, Dc, 1.0f);

  qrel_kernel<<<(Bc * Hc * Lc) / 128, 256, 0, stream>>>(qh, relbf, qrel);

  attn_kernel<<<Bc * Hc * (Lc / 128), 256, 0, stream>>>(qh, kh, vh, qrel, relbf,
                                                        mask, ctx);

  gemm_bf16_kernel<1><<<ggrid, 256, 0, stream>>>(ctx, wo, bo, out, nullptr,
                                                 Mrows, Dc, Dc, 1.0f);
}
